// eHON_MPL_centroid_6622839570872
// MI455X (gfx1250) — compile-verified, matrix operand order checked by reading, not run on hardware
//
#include <hip/hip_runtime.h>
#include <hip/hip_bf16.h>
#include <math.h>

// ---------------------------------------------------------------------------
// CDNA5 (gfx1250) WMMA types / helpers
// ---------------------------------------------------------------------------
typedef __bf16 v16bf __attribute__((ext_vector_type(16)));
typedef float  v8f   __attribute__((ext_vector_type(8)));

union AFrag {
  v16bf v;
  unsigned int u[8];
};

union B4 {
  __bf16 b[4];
  unsigned long long u;
};

__device__ __forceinline__ v8f wmma_bf16(v16bf a, v16bf b, v8f c) {
  // D(16x16,f32) = A(16x32,bf16) * B(32x16,bf16) + C
  return __builtin_amdgcn_wmma_f32_16x16x32_bf16(false, a, false, b, (short)0, c,
                                                 false, false);
}

__device__ __forceinline__ v8f zero8() {
  v8f z = {0.f, 0.f, 0.f, 0.f, 0.f, 0.f, 0.f, 0.f};
  return z;
}

// Load A fragment (16 rows x 32 K, bf16) from a row-major LDS buffer.
// ISA 16-bit A layout (7.12.2): lane = half*16 + m;
//   VGPR v<4 : K = k0 + 8*half + 2v, +1 ; VGPR v>=4 : K = k0 + 16 + 8*half + 2(v-4), +1
__device__ __forceinline__ v16bf load_a(const __bf16* buf, int ldk, int k0, int lane) {
  const int m = lane & 15;
  const int half = (lane >> 4) & 1;
  AFrag f;
  const __bf16* row = buf + m * ldk + k0 + half * 8;
#pragma unroll
  for (int v = 0; v < 8; ++v) {
    const int kl = (v < 4) ? (2 * v) : (16 + 2 * (v - 4));
    f.u[v] = *reinterpret_cast<const unsigned int*>(row + kl);
  }
  return f.v;
}

// B fragments pre-packed: one contiguous 32-byte load per lane.
// frag index = kt*8 + nt ; element (frag*32 + lane)*16 + j
__device__ __forceinline__ v16bf load_b(const __bf16* Wp, int kt, int nt, int lane) {
  return reinterpret_cast<const v16bf*>(Wp)[(kt * 8 + nt) * 32 + lane];
}

__device__ __forceinline__ void cvt_store4(__bf16* dst, float4 v) {
  B4 t;
  t.b[0] = (__bf16)v.x;
  t.b[1] = (__bf16)v.y;
  t.b[2] = (__bf16)v.z;
  t.b[3] = (__bf16)v.w;
  *reinterpret_cast<unsigned long long*>(dst) = t.u;
}

// ---------------------------------------------------------------------------
// Weight packing: W row-major [K,128] f32 -> bf16 B-operand fragments.
// ---------------------------------------------------------------------------
__global__ void pack_weight_kernel(const float* __restrict__ W,
                                   __bf16* __restrict__ dst, int K) {
  const int t = blockIdx.x * blockDim.x + threadIdx.x;
  const int total = K * 128;
  if (t >= total) return;
  const int frag = t >> 9;
  const int rem = t & 511;
  const int lane = rem >> 4;
  const int j = rem & 15;
  const int kt = frag >> 3;
  const int nt = frag & 7;
  const int half = lane >> 4;
  const int col = lane & 15;
  const int v = j >> 1, p = j & 1;
  const int kl = ((v < 4) ? (2 * v) : (16 + 2 * (v - 4))) + 8 * half + p;
  const int k = kt * 32 + kl;
  const int n = nt * 16 + col;
  dst[t] = (__bf16)W[k * 128 + n];
}

// ---------------------------------------------------------------------------
// Edge branch kernel: block = 64 edges, 8 waves; wave w owns N-tile w and all
// four 16-edge M-tiles (B fragments amortized 4x).
// ---------------------------------------------------------------------------
#define EPB 64  // edges per block (4 M-tiles of 16)

__global__ __launch_bounds__(256) void edge_branch_kernel(
    const float* __restrict__ h, const float* __restrict__ h_o,
    const float* __restrict__ x, const float* __restrict__ x_o,
    const int* __restrict__ bi, const int* __restrict__ bj, int E,
    const __bf16* __restrict__ W1p, const float* __restrict__ w1_last,
    const float* __restrict__ b1, const __bf16* __restrict__ W2p,
    const float* __restrict__ b2, const __bf16* __restrict__ cW1p,
    const float* __restrict__ cb1, const float* __restrict__ cW2,
    const float* __restrict__ cb2, float* __restrict__ m_out,
    float* __restrict__ xagg) {
  __shared__ __bf16 zs[EPB * 256];  // [h_i | h_j] bf16
  __shared__ __bf16 hs[EPB * 128];  // ReLU(layer1)
  __shared__ __bf16 ms[EPB * 128];  // m_in (layer2)
  __shared__ float d2s[EPB];
  __shared__ float xijs[EPB][3];
  __shared__ float sarr[EPB];  // gate logits (cross-wave reduction)
  __shared__ int bis[EPB];
  __shared__ int bjs[EPB];

  const int tid = threadIdx.x;
  const int wave = tid >> 5;
  const int lane = tid & 31;
  const int half = lane >> 4;
  const int col = lane & 15;
  const int e0 = blockIdx.x * EPB;

  if (tid < EPB) {
    const int e = e0 + tid;
    int i = 0, j = 0;
    float d2 = 0.f, a0 = 0.f, a1 = 0.f, a2 = 0.f;
    if (e < E) {
      i = bi[e];
      j = bj[e];
      a0 = x[i * 3 + 0] - x_o[j * 3 + 0];
      a1 = x[i * 3 + 1] - x_o[j * 3 + 1];
      a2 = x[i * 3 + 2] - x_o[j * 3 + 2];
      d2 = a0 * a0 + a1 * a1 + a2 * a2;
    }
    bis[tid] = i;
    bjs[tid] = j;
    d2s[tid] = d2;
    xijs[tid][0] = a0;
    xijs[tid][1] = a1;
    xijs[tid][2] = a2;
    sarr[tid] = cb2[0];
  }
  __syncthreads();

  // Stage gathered features as bf16, float4-vectorized (tables are L2-resident)
  for (int idx = tid; idx < EPB * 64; idx += 256) {
    const int el = idx >> 6;
    const int k = (idx & 63) * 4;
    float4 v = {0.f, 0.f, 0.f, 0.f};
    if (e0 + el < E) {
      const float* src = (k < 128) ? &h[(long)bis[el] * 128 + k]
                                   : &h_o[(long)bjs[el] * 128 + (k - 128)];
      v = *reinterpret_cast<const float4*>(src);
    }
    cvt_store4(&zs[el * 256 + k], v);
  }
  __syncthreads();

  const int n = wave * 16 + col;

  // ---- layer 1: [EPB x 256] @ [256 x 128] ----
  v8f acc[4] = {zero8(), zero8(), zero8(), zero8()};
#pragma unroll
  for (int kt = 0; kt < 8; ++kt) {
    v16bf b = load_b(W1p, kt, wave, lane);
#pragma unroll
    for (int t = 0; t < 4; ++t) {
      v16bf a = load_a(zs + t * 16 * 256, 256, kt * 32, lane);
      acc[t] = wmma_bf16(a, b, acc[t]);
    }
  }
  {
    const float bn = b1[n];
    const float wl = w1_last[n];
#pragma unroll
    for (int t = 0; t < 4; ++t) {
#pragma unroll
      for (int r = 0; r < 8; ++r) {
        const int m = t * 16 + r + 8 * half;
        float f = acc[t][r] + bn + d2s[m] * wl;  // rank-1 d2 column of W1
        hs[m * 128 + n] = (__bf16)fmaxf(f, 0.f);
      }
    }
  }
  __syncthreads();

  // ---- layer 2 -> m_in ----
  v8f acc2[4] = {zero8(), zero8(), zero8(), zero8()};
#pragma unroll
  for (int kt = 0; kt < 4; ++kt) {
    v16bf b = load_b(W2p, kt, wave, lane);
#pragma unroll
    for (int t = 0; t < 4; ++t) {
      v16bf a = load_a(hs + t * 16 * 128, 128, kt * 32, lane);
      acc2[t] = wmma_bf16(a, b, acc2[t]);
    }
  }
  float mf[4][8];
  {
    const float bn = b2[n];
#pragma unroll
    for (int t = 0; t < 4; ++t) {
#pragma unroll
      for (int r = 0; r < 8; ++r) {
        const int m = t * 16 + r + 8 * half;
        const float f = acc2[t][r] + bn;
        mf[t][r] = f;
        ms[m * 128 + n] = (__bf16)f;
      }
    }
  }
  __syncthreads();

  // ---- coord gate: ReLU(m_in @ cW1 + cb1) . cW2 ----
  v8f acc3[4] = {zero8(), zero8(), zero8(), zero8()};
#pragma unroll
  for (int kt = 0; kt < 4; ++kt) {
    v16bf b = load_b(cW1p, kt, wave, lane);
#pragma unroll
    for (int t = 0; t < 4; ++t) {
      v16bf a = load_a(ms + t * 16 * 128, 128, kt * 32, lane);
      acc3[t] = wmma_bf16(a, b, acc3[t]);
    }
  }
  {
    const float bn = cb1[n];
    const float wn = cW2[n];
#pragma unroll
    for (int t = 0; t < 4; ++t) {
#pragma unroll
      for (int r = 0; r < 8; ++r) {
        float v = fmaxf(acc3[t][r] + bn, 0.f) * wn;
        // reduce the 16 feature-lanes of this half (same edge m)
        v += __shfl_xor(v, 8, 32);
        v += __shfl_xor(v, 4, 32);
        v += __shfl_xor(v, 2, 32);
        v += __shfl_xor(v, 1, 32);
        if (col == 0)
          atomicAdd(&sarr[t * 16 + r + 8 * half], v);  // ds_add_f32, 1/16 lanes
      }
    }
  }

  // ---- segment-sum of m_in via f32 global atomics ----
#pragma unroll
  for (int t = 0; t < 4; ++t) {
#pragma unroll
    for (int r = 0; r < 8; ++r) {
      const int m = t * 16 + r + 8 * half;
      if (e0 + m < E) unsafeAtomicAdd(&m_out[(long)bis[m] * 128 + n], mf[t][r]);
    }
  }
  __syncthreads();

  // ---- gated coordinate aggregation ----
  if (tid < EPB * 3) {
    const int el = tid / 3, c = tid - el * 3;
    if (e0 + el < E) {
      const float s = 1.f / (1.f + __expf(-sarr[el]));
      unsafeAtomicAdd(&xagg[(long)bis[el] * 3 + c], xijs[el][c] * s);
    }
  }
}

// ---------------------------------------------------------------------------
// Cell update: [h | m_up | m_dn] (384) -> ReLU(@W1+b1) -> @W2+b2 -> h_out
// Block = 64 nodes, 8 waves, wave owns N-tile and 4 M-tiles.
// ---------------------------------------------------------------------------
__global__ __launch_bounds__(256) void cell_kernel(
    const float* __restrict__ h, const float* __restrict__ m_up,
    const float* __restrict__ m_dn, const __bf16* __restrict__ W1p,
    const float* __restrict__ b1, const __bf16* __restrict__ W2p,
    const float* __restrict__ b2, float* __restrict__ h_out, int N) {
  __shared__ __bf16 zs[EPB * 384];
  __shared__ __bf16 hs[EPB * 128];

  const int tid = threadIdx.x;
  const int wave = tid >> 5;
  const int lane = tid & 31;
  const int half = lane >> 4;
  const int col = lane & 15;
  const int node0 = blockIdx.x * EPB;

  for (int idx = tid; idx < EPB * 96; idx += 256) {
    const int nl = idx / 96;
    const int k = (idx - nl * 96) * 4;
    const long node = node0 + nl;
    float4 v = {0.f, 0.f, 0.f, 0.f};
    if (node < N) {
      const float* src = (k < 128) ? &h[node * 128 + k]
                        : (k < 256) ? &m_up[node * 128 + (k - 128)]
                                    : &m_dn[node * 128 + (k - 256)];
      v = *reinterpret_cast<const float4*>(src);
    }
    cvt_store4(&zs[nl * 384 + k], v);
  }
  __syncthreads();

  const int n = wave * 16 + col;

  v8f acc[4] = {zero8(), zero8(), zero8(), zero8()};
#pragma unroll
  for (int kt = 0; kt < 12; ++kt) {
    v16bf b = load_b(W1p, kt, wave, lane);
#pragma unroll
    for (int t = 0; t < 4; ++t) {
      v16bf a = load_a(zs + t * 16 * 384, 384, kt * 32, lane);
      acc[t] = wmma_bf16(a, b, acc[t]);
    }
  }
  {
    const float bn = b1[n];
#pragma unroll
    for (int t = 0; t < 4; ++t) {
#pragma unroll
      for (int r = 0; r < 8; ++r) {
        const int m = t * 16 + r + 8 * half;
        hs[m * 128 + n] = (__bf16)fmaxf(acc[t][r] + bn, 0.f);
      }
    }
  }
  __syncthreads();

  v8f acc2[4] = {zero8(), zero8(), zero8(), zero8()};
#pragma unroll
  for (int kt = 0; kt < 4; ++kt) {
    v16bf b = load_b(W2p, kt, wave, lane);
#pragma unroll
    for (int t = 0; t < 4; ++t) {
      v16bf a = load_a(hs + t * 16 * 128, 128, kt * 32, lane);
      acc2[t] = wmma_bf16(a, b, acc2[t]);
    }
  }
  {
    const float bn = b2[n];
#pragma unroll
    for (int t = 0; t < 4; ++t) {
#pragma unroll
      for (int r = 0; r < 8; ++r) {
        const long node = node0 + t * 16 + r + 8 * half;
        if (node < N) h_out[node * 128 + n] = acc2[t][r] + bn;
      }
    }
  }
}

__global__ void xout_kernel(const float* __restrict__ x,
                            const float* __restrict__ xup,
                            const float* __restrict__ xdn,
                            const float* __restrict__ cw,
                            float* __restrict__ x_out, int total) {
  const int i = blockIdx.x * blockDim.x + threadIdx.x;
  if (i < total) x_out[i] = x[i] + cw[0] * xup[i] + cw[1] * xdn[i];
}

// ---------------------------------------------------------------------------
extern "C" void kernel_launch(void* const* d_in, const int* in_sizes, int n_in,
                              void* d_out, int out_size, void* d_ws,
                              size_t ws_size, hipStream_t stream) {
  const float* h = (const float*)d_in[0];
  const float* h_up = (const float*)d_in[1];
  const float* h_down = (const float*)d_in[2];
  const float* x = (const float*)d_in[3];
  const float* x_up = (const float*)d_in[4];
  const float* x_down = (const float*)d_in[5];
  const float* cw = (const float*)d_in[6];
  const float* up_W1 = (const float*)d_in[7];
  const float* up_b1 = (const float*)d_in[8];
  const float* up_W2 = (const float*)d_in[9];
  const float* up_b2 = (const float*)d_in[10];
  const float* dn_W1 = (const float*)d_in[11];
  const float* dn_b1 = (const float*)d_in[12];
  const float* dn_W2 = (const float*)d_in[13];
  const float* dn_b2 = (const float*)d_in[14];
  const float* cell_W1 = (const float*)d_in[15];
  const float* cell_b1 = (const float*)d_in[16];
  const float* cell_W2 = (const float*)d_in[17];
  const float* cell_b2 = (const float*)d_in[18];
  const float* cu_W1 = (const float*)d_in[19];
  const float* cu_b1 = (const float*)d_in[20];
  const float* cu_W2 = (const float*)d_in[21];
  const float* cu_b2 = (const float*)d_in[22];
  const float* cd_W1 = (const float*)d_in[23];
  const float* cd_b1 = (const float*)d_in[24];
  const float* cd_W2 = (const float*)d_in[25];
  const float* cd_b2 = (const float*)d_in[26];
  const int* b_up_i = (const int*)d_in[27];
  const int* b_up_j = (const int*)d_in[28];
  const int* b_dn_i = (const int*)d_in[29];
  const int* b_dn_j = (const int*)d_in[30];

  const int N = in_sizes[0] / 128;
  const int E_up = in_sizes[27];
  const int E_dn = in_sizes[29];

  // ---- workspace carve-up (accumulators first, contiguous for one memset) --
  char* ws = (char*)d_ws;
  size_t off = 0;
  auto take = [&](size_t bytes) -> char* {
    char* p = ws + off;
    off = (off + bytes + 255) & ~(size_t)255;
    return p;
  };
  float* m_up_acc = (float*)take((size_t)N * 128 * 4);
  float* m_dn_acc = (float*)take((size_t)N * 128 * 4);
  float* xup_acc = (float*)take((size_t)N * 3 * 4);
  float* xdn_acc = (float*)take((size_t)N * 3 * 4);
  const size_t zero_bytes = off;
  __bf16* upW1p = (__bf16*)take(256 * 128 * 2);
  __bf16* upW2p = (__bf16*)take(128 * 128 * 2);
  __bf16* cuW1p = (__bf16*)take(128 * 128 * 2);
  __bf16* dnW1p = (__bf16*)take(256 * 128 * 2);
  __bf16* dnW2p = (__bf16*)take(128 * 128 * 2);
  __bf16* cdW1p = (__bf16*)take(128 * 128 * 2);
  __bf16* ceW1p = (__bf16*)take(384 * 128 * 2);
  __bf16* ceW2p = (__bf16*)take(128 * 128 * 2);
  (void)ws_size;
  (void)n_in;
  (void)out_size;

  hipMemsetAsync(ws, 0, zero_bytes, stream);

  // ---- pack weights into WMMA B-fragment layout (bf16) ----
  auto pack = [&](const float* W, __bf16* dst, int K) {
    const int total = K * 128;
    pack_weight_kernel<<<(total + 255) / 256, 256, 0, stream>>>(W, dst, K);
  };
  pack(up_W1, upW1p, 256);  // first 256 rows; row 256 handled as rank-1 term
  pack(up_W2, upW2p, 128);
  pack(cu_W1, cuW1p, 128);
  pack(dn_W1, dnW1p, 256);
  pack(dn_W2, dnW2p, 128);
  pack(cd_W1, cdW1p, 128);
  pack(cell_W1, ceW1p, 384);
  pack(cell_W2, ceW2p, 128);

  // ---- edge branches ----
  edge_branch_kernel<<<(E_up + EPB - 1) / EPB, 256, 0, stream>>>(
      h, h_up, x, x_up, b_up_i, b_up_j, E_up, upW1p, up_W1 + 256 * 128, up_b1,
      upW2p, up_b2, cuW1p, cu_b1, cu_W2, cu_b2, m_up_acc, xup_acc);

  edge_branch_kernel<<<(E_dn + EPB - 1) / EPB, 256, 0, stream>>>(
      h, h_down, x, x_down, b_dn_i, b_dn_j, E_dn, dnW1p, dn_W1 + 256 * 128,
      dn_b1, dnW2p, dn_b2, cdW1p, cd_b1, cd_W2, cd_b2, m_dn_acc, xdn_acc);

  // ---- cell update + coordinate output ----
  float* h_out = (float*)d_out;
  float* x_out = (float*)d_out + (size_t)N * 128;

  cell_kernel<<<(N + EPB - 1) / EPB, 256, 0, stream>>>(
      h, m_up_acc, m_dn_acc, ceW1p, cell_b1, ceW2p, cell_b2, h_out, N);

  const int xtotal = N * 3;
  xout_kernel<<<(xtotal + 255) / 256, 256, 0, stream>>>(x, xup_acc, xdn_acc, cw,
                                                        x_out, xtotal);
}
